// T5Attention_50672024158946
// MI455X (gfx1250) — matrix operand order repaired and checked by reading, not compiled, not run on hardware
//
#include <hip/hip_runtime.h>
#include <hip/hip_bf16.h>
#include <math.h>
#include <stdint.h>

#define S_LEN   2048
#define D_MODEL 1024
#define NH      16
#define DKV     64
#define BATCH   2
#define M_TOTAL (BATCH * S_LEN)   // 4096

typedef __attribute__((ext_vector_type(16))) __bf16 v16bf;
typedef __attribute__((ext_vector_type(8)))  float  v8f;

union Frag16 { v16bf v; unsigned short u[16]; };

__device__ __forceinline__ unsigned short f2bf(float x) {
  unsigned u = __builtin_bit_cast(unsigned, x);
  u += 0x7fffu + ((u >> 16) & 1u);
  return (unsigned short)(u >> 16);
}

// Async global->LDS copy, 16 bytes per lane. Tracked with ASYNCcnt.
// lds_off: wave-relative LDS byte address (low 32 bits of a __shared__ ptr).
__device__ __forceinline__ void async_load_b128(unsigned lds_off, const void* gaddr) {
  unsigned long long ga = (unsigned long long)(uintptr_t)gaddr;
  asm volatile("global_load_async_to_lds_b128 %0, %1, off"
               :: "v"(lds_off), "v"(ga)
               : "memory");
}

#define WAIT_ASYNC(n) asm volatile("s_wait_asynccnt " #n ::: "memory")

// A-matrix 16x32 bf16 fragment from row-major [16][ld] (K contiguous).
// Layout: lane row = lane&15; element s holds k = 16*half + s (contiguous order).
__device__ __forceinline__ v16bf load_a16(const unsigned short* src, int ld) {
  const int lane = threadIdx.x & 31;
  const int row  = lane & 15;
  const int half = lane >> 4;
  Frag16 f;
#pragma unroll
  for (int v = 0; v < 8; ++v) {
    int kb = half * 16 + 2 * v;
    f.u[2 * v]     = src[row * ld + kb];
    f.u[2 * v + 1] = src[row * ld + kb + 1];
  }
  return f.v;
}

// B-matrix 32x16 bf16 fragment where B[k][n] = src[n*ld + k]
__device__ __forceinline__ v16bf load_b16_nk(const unsigned short* src, int ld) {
  const int lane = threadIdx.x & 31;
  const int n    = lane & 15;
  const int half = lane >> 4;
  Frag16 f;
#pragma unroll
  for (int v = 0; v < 8; ++v) {
    int k = half * 16 + 2 * v;
    f.u[2 * v]     = src[n * ld + k];
    f.u[2 * v + 1] = src[n * ld + k + 1];
  }
  return f.v;
}

// B-matrix 32x16 bf16 fragment where B[k][n] = src[k*ld + n0 + (lane&15)]
__device__ __forceinline__ v16bf load_b16_kn(const unsigned short* src, int ld, int n0) {
  const int lane = threadIdx.x & 31;
  const int n    = n0 + (lane & 15);
  const int half = lane >> 4;
  Frag16 f;
#pragma unroll
  for (int v = 0; v < 8; ++v) {
    int k = half * 16 + 2 * v;
    f.u[2 * v]     = src[k * ld + n];
    f.u[2 * v + 1] = src[(k + 1) * ld + n];
  }
  return f.v;
}

__device__ __forceinline__ v8f wmma_bf16(v16bf a, v16bf b, v8f c) {
  return __builtin_amdgcn_wmma_f32_16x16x32_bf16(false, a, false, b, (short)0, c,
                                                 false, false);
}

// T5 relative position bucket (bidirectional, 32 buckets, max_distance 128)
__device__ __forceinline__ int rel_bucket(int rel) {
  int rb = (rel > 0) ? 16 : 0;
  int ap = rel < 0 ? -rel : rel;
  if (ap < 8) return rb + ap;
  int lg = 8 + (int)(__logf((float)ap * 0.125f) * 2.8853900817779268f);
  lg = lg > 15 ? 15 : lg;
  return rb + lg;
}

__global__ void f32_to_bf16_kernel(const float* __restrict__ in,
                                   unsigned short* __restrict__ out, int n) {
  int i = blockIdx.x * 256 + threadIdx.x;
  if (i < n) out[i] = f2bf(in[i]);
}

// ---------------------------------------------------------------------------
// GEMM: block computes 128x64 of out = A @ W^T. 4 waves; wave w covers rows
// [w*32, w*32+32). Per 32-wide k-step: 2 A-frags x 4 B-frags -> 8 WMMAs.
// Tiles double-buffered via global_load_async_to_lds_b128: tile i+1 streams
// in (ASYNCcnt) while tile i feeds the WMMA pipes.
// LDS pad: ld=40 ushorts -> 80B rows, every 16B slot aligned.
// ---------------------------------------------------------------------------
#define LDT 40

// out scattered into [B,H,S,DKV] bf16; W selected by blockIdx.z in {Wq,Wk,Wv}.
__global__ __launch_bounds__(128) void gemm_qkv_bf16(
    const unsigned short* __restrict__ X,   // [4096][1024] bf16
    const unsigned short* __restrict__ Wq,  // [1024][1024] bf16
    const unsigned short* __restrict__ Wk,
    const unsigned short* __restrict__ Wv,
    unsigned short* __restrict__ QKV) {     // 3 x [B][H][S][DKV]
  __shared__ unsigned short Xs[2][128 * LDT];
  __shared__ unsigned short Ws[2][64 * LDT];
  const int t = threadIdx.x;
  const int w = t >> 5;
  const int lane = t & 31;
  const int half = lane >> 4;
  const int nlane = lane & 15;
  const int n0 = blockIdx.x * 64;
  const int m0 = blockIdx.y * 128;
  const unsigned short* W = (blockIdx.z == 0) ? Wq : (blockIdx.z == 1) ? Wk : Wv;
  unsigned short* out = QKV + (size_t)blockIdx.z * (size_t)(M_TOTAL * D_MODEL);

  // 6 async-load instructions per tile issue (4 X + 2 W)
  auto issue_tile = [&](int kk, int buf) {
#pragma unroll
    for (int i = 0; i < 4; ++i) {
      int e = t + i * 128;            // X: 128 rows x 32 cols = 512 b128 units
      int row = e >> 2;
      int c8 = (e & 3) * 8;
      async_load_b128((unsigned)(uintptr_t)&Xs[buf][row * LDT + c8],
                      &X[(m0 + row) * D_MODEL + kk + c8]);
    }
#pragma unroll
    for (int i = 0; i < 2; ++i) {
      int e = t + i * 128;            // W: 64 rows x 32 cols = 256 b128 units
      int row = e >> 2;
      int c8 = (e & 3) * 8;
      async_load_b128((unsigned)(uintptr_t)&Ws[buf][row * LDT + c8],
                      &W[(n0 + row) * D_MODEL + kk + c8]);
    }
  };

  v8f c[2][4] = {};
  issue_tile(0, 0);
  for (int kk = 0, it = 0; kk < D_MODEL; kk += 32, ++it) {
    const int cur = it & 1;
    __syncthreads();                  // readers of buf cur^1 are done
    if (kk + 32 < D_MODEL) {
      issue_tile(kk + 32, cur ^ 1);   // stream next tile behind compute
      WAIT_ASYNC(0x6);                // drain only tile `cur`'s 6 ops
    } else {
      WAIT_ASYNC(0x0);
    }
    __syncthreads();

    v16bf a0 = load_a16(&Xs[cur][(w * 32 + 0) * LDT], LDT);
    v16bf a1 = load_a16(&Xs[cur][(w * 32 + 16) * LDT], LDT);
#pragma unroll
    for (int nt = 0; nt < 4; ++nt) {
      v16bf b = load_b16_nk(&Ws[cur][nt * 16 * LDT], LDT);
      c[0][nt] = wmma_bf16(a0, b, c[0][nt]);
      c[1][nt] = wmma_bf16(a1, b, c[1][nt]);
    }
  }
#pragma unroll
  for (int mt = 0; mt < 2; ++mt) {
#pragma unroll
    for (int nt = 0; nt < 4; ++nt) {
#pragma unroll
      for (int r = 0; r < 8; ++r) {
        int m = m0 + w * 32 + mt * 16 + r + 8 * half;
        int n = n0 + nt * 16 + nlane;
        int b = m >> 11, s = m & 2047;
        int h = n >> 6,  d = n & 63;
        out[(((size_t)(b * NH + h) * S_LEN) + s) * DKV + d] = f2bf(c[mt][nt][r]);
      }
    }
  }
}

// ---------------------------------------------------------------------------
// Flash attention with T5 relative position bias. Block = 4 waves sharing a
// 32-key K/V LDS tile; tiles double-buffered with async global->LDS DMA.
// Each wave owns 16 query rows.
// ---------------------------------------------------------------------------
__global__ __launch_bounds__(128) void flash_attn_bf16(
    const unsigned short* __restrict__ Q,   // [B][H][S][DKV] bf16
    const unsigned short* __restrict__ K,
    const unsigned short* __restrict__ V,
    const float* __restrict__ rel_emb,      // [32][NH] fp32
    unsigned short* __restrict__ CTX) {     // [B][S][H*DKV] bf16
  __shared__ unsigned short Ks[2][32 * 72]; // 144B rows: b128-aligned
  __shared__ unsigned short Vs[2][32 * 72];
  __shared__ unsigned short Ps[4 * 16 * 34];
  __shared__ float bias[32];

  const int t = threadIdx.x;
  const int w = t >> 5;
  const int lane = t & 31;
  const int half = lane >> 4;
  const int nlane = lane & 15;
  const int h = blockIdx.y;
  const int b = blockIdx.z;
  const int q0 = blockIdx.x * 64 + w * 16;

  if (t < 32) bias[t] = rel_emb[t * NH + h];

  const size_t head_off = (size_t)((b * NH + h) * S_LEN) * DKV;
  const unsigned short* Qh = Q + head_off;
  const unsigned short* Kh = K + head_off;
  const unsigned short* Vh = V + head_off;

  // 4 async-load instructions per tile issue (2 K + 2 V)
  auto issue_kv = [&](int kt, int buf) {
#pragma unroll
    for (int i = 0; i < 2; ++i) {
      int e = t + i * 128;            // 32 rows x 64 cols = 256 b128 units
      int row = e >> 3;
      int c8 = (e & 7) * 8;
      async_load_b128((unsigned)(uintptr_t)&Ks[buf][row * 72 + c8],
                      &Kh[(kt + row) * DKV + c8]);
      async_load_b128((unsigned)(uintptr_t)&Vs[buf][row * 72 + c8],
                      &Vh[(kt + row) * DKV + c8]);
    }
  };

  // Q A-fragments for d=0..31 and d=32..63, resident for the whole pass
  v16bf aq0 = load_a16(&Qh[q0 * DKV], DKV);
  v16bf aq1 = load_a16(&Qh[q0 * DKV + 32], DKV);

  float mrun[8], lrun[8];
  v8f acc0 = {}, acc1 = {}, acc2 = {}, acc3 = {};
#pragma unroll
  for (int r = 0; r < 8; ++r) { mrun[r] = -3.0e38f; lrun[r] = 0.f; }

  issue_kv(0, 0);
  for (int kt = 0, it = 0; kt < S_LEN; kt += 32, ++it) {
    const int cur = it & 1;
    __syncthreads();                  // readers of buf cur^1 are done
    if (kt + 32 < S_LEN) {
      issue_kv(kt + 32, cur ^ 1);     // stream next K/V tile behind compute
      WAIT_ASYNC(0x4);                // drain only tile `cur`'s 4 ops
    } else {
      WAIT_ASYNC(0x0);
    }
    __syncthreads();
    const unsigned short* Kc = Ks[cur];
    const unsigned short* Vc = Vs[cur];

    // scores: S = Q K^T  (two 16-key tiles, dk=64 -> 2 wmma steps each)
    v8f s0 = {}, s1 = {};
    {
      v16bf bk = load_b16_nk(&Kc[0], 72);
      s0 = wmma_bf16(aq0, bk, s0);
      bk = load_b16_nk(&Kc[32], 72);
      s0 = wmma_bf16(aq1, bk, s0);
      bk = load_b16_nk(&Kc[16 * 72], 72);
      s1 = wmma_bf16(aq0, bk, s1);
      bk = load_b16_nk(&Kc[16 * 72 + 32], 72);
      s1 = wmma_bf16(aq1, bk, s1);
    }
    // add T5 relative position bias
#pragma unroll
    for (int r = 0; r < 8; ++r) {
      int qrow = q0 + r + 8 * half;
      int kpos = kt + nlane;
      s0[r] += bias[rel_bucket(kpos - qrow)];
      s1[r] += bias[rel_bucket(kpos + 16 - qrow)];
    }
    // online softmax (row lives in one 16-lane half; xor-reduce within it)
    unsigned short* pw = &Ps[w * 16 * 34];
    float alpha[8];
#pragma unroll
    for (int r = 0; r < 8; ++r) {
      float tm = fmaxf(s0[r], s1[r]);
      tm = fmaxf(tm, __shfl_xor(tm, 1, 32));
      tm = fmaxf(tm, __shfl_xor(tm, 2, 32));
      tm = fmaxf(tm, __shfl_xor(tm, 4, 32));
      tm = fmaxf(tm, __shfl_xor(tm, 8, 32));
      float nm = fmaxf(mrun[r], tm);
      alpha[r] = __expf(mrun[r] - nm);
      mrun[r] = nm;
      float p0 = __expf(s0[r] - nm);
      float p1 = __expf(s1[r] - nm);
      float rs = p0 + p1;
      rs += __shfl_xor(rs, 1, 32);
      rs += __shfl_xor(rs, 2, 32);
      rs += __shfl_xor(rs, 4, 32);
      rs += __shfl_xor(rs, 8, 32);
      lrun[r] = lrun[r] * alpha[r] + rs;
      pw[(r + 8 * half) * 34 + nlane]      = f2bf(p0);
      pw[(r + 8 * half) * 34 + 16 + nlane] = f2bf(p1);
      acc0[r] *= alpha[r];
      acc1[r] *= alpha[r];
      acc2[r] *= alpha[r];
      acc3[r] *= alpha[r];
    }
    // P store->load is same-wave LDS (in-order per ISA); no barrier needed.

    // PV: acc += P V
    v16bf pa = load_a16(pw, 34);
    v16bf bv = load_b16_kn(Vc, 72, 0);
    acc0 = wmma_bf16(pa, bv, acc0);
    bv = load_b16_kn(Vc, 72, 16);
    acc1 = wmma_bf16(pa, bv, acc1);
    bv = load_b16_kn(Vc, 72, 32);
    acc2 = wmma_bf16(pa, bv, acc2);
    bv = load_b16_kn(Vc, 72, 48);
    acc3 = wmma_bf16(pa, bv, acc3);
  }

  // epilogue: ctx[b][s][h*64 + d] = acc / l
#pragma unroll
  for (int r = 0; r < 8; ++r) {
    float inv = 1.0f / lrun[r];
    int srow = q0 + r + 8 * half;
    size_t o = ((size_t)(b * S_LEN + srow)) * D_MODEL + h * DKV + nlane;
    CTX[o + 0]  = f2bf(acc0[r] * inv);
    CTX[o + 16] = f2bf(acc1[r] * inv);
    CTX[o + 32] = f2bf(acc2[r] * inv);
    CTX[o + 48] = f2bf(acc3[r] * inv);
  }
}

// out = ctx @ Wo^T, fp32 output, 128x64 block tiles, double-buffered
__global__ __launch_bounds__(128) void gemm_out_f32(
    const unsigned short* __restrict__ A,   // [4096][1024] bf16
    const unsigned short* __restrict__ Wo,  // [1024][1024] bf16
    float* __restrict__ out) {              // [4096][1024] fp32
  __shared__ unsigned short Xs[2][128 * LDT];
  __shared__ unsigned short Ws[2][64 * LDT];
  const int t = threadIdx.x;
  const int w = t >> 5;
  const int lane = t & 31;
  const int half = lane >> 4;
  const int nlane = lane & 15;
  const int n0 = blockIdx.x * 64;
  const int m0 = blockIdx.y * 128;

  auto issue_tile = [&](int kk, int buf) {
#pragma unroll
    for (int i = 0; i < 4; ++i) {
      int e = t + i * 128;
      int row = e >> 2;
      int c8 = (e & 3) * 8;
      async_load_b128((unsigned)(uintptr_t)&Xs[buf][row * LDT + c8],
                      &A[(m0 + row) * D_MODEL + kk + c8]);
    }
#pragma unroll
    for (int i = 0; i < 2; ++i) {
      int e = t + i * 128;
      int row = e >> 2;
      int c8 = (e & 3) * 8;
      async_load_b128((unsigned)(uintptr_t)&Ws[buf][row * LDT + c8],
                      &Wo[(n0 + row) * D_MODEL + kk + c8]);
    }
  };

  v8f c[2][4] = {};
  issue_tile(0, 0);
  for (int kk = 0, it = 0; kk < D_MODEL; kk += 32, ++it) {
    const int cur = it & 1;
    __syncthreads();
    if (kk + 32 < D_MODEL) {
      issue_tile(kk + 32, cur ^ 1);
      WAIT_ASYNC(0x6);
    } else {
      WAIT_ASYNC(0x0);
    }
    __syncthreads();

    v16bf a0 = load_a16(&Xs[cur][(w * 32 + 0) * LDT], LDT);
    v16bf a1 = load_a16(&Xs[cur][(w * 32 + 16) * LDT], LDT);
#pragma unroll
    for (int nt = 0; nt < 4; ++nt) {
      v16bf b = load_b16_nk(&Ws[cur][nt * 16 * LDT], LDT);
      c[0][nt] = wmma_bf16(a0, b, c[0][nt]);
      c[1][nt] = wmma_bf16(a1, b, c[1][nt]);
    }
  }
#pragma unroll
  for (int mt = 0; mt < 2; ++mt) {
#pragma unroll
    for (int nt = 0; nt < 4; ++nt) {
#pragma unroll
      for (int r = 0; r < 8; ++r) {
        int m = m0 + w * 32 + mt * 16 + r + 8 * half;
        int n = n0 + nt * 16 + nlane;
        out[(size_t)m * D_MODEL + n] = c[mt][nt][r];
      }
    }
  }
}

extern "C" void kernel_launch(void* const* d_in, const int* in_sizes, int n_in,
                              void* d_out, int out_size, void* d_ws, size_t ws_size,
                              hipStream_t stream) {
  (void)in_sizes; (void)n_in; (void)out_size; (void)ws_size;
  const float* hs  = (const float*)d_in[0];
  const float* Wq  = (const float*)d_in[1];
  const float* Wk  = (const float*)d_in[2];
  const float* Wv  = (const float*)d_in[3];
  const float* Wo  = (const float*)d_in[4];
  const float* rel = (const float*)d_in[5];

  unsigned short* ws = (unsigned short*)d_ws;
  const size_t NX = (size_t)M_TOTAL * D_MODEL;   // 4,194,304
  const size_t NW = (size_t)D_MODEL * D_MODEL;   // 1,048,576
  unsigned short* Xb   = ws;
  unsigned short* Wqb  = Xb + NX;
  unsigned short* Wkb  = Wqb + NW;
  unsigned short* Wvb  = Wkb + NW;
  unsigned short* Wob  = Wvb + NW;
  unsigned short* QKVb = Wob + NW;   // 3 * NX
  unsigned short* Qb   = QKVb;
  unsigned short* Kb   = Qb + NX;
  unsigned short* Vb   = Kb + NX;
  unsigned short* Cb   = Vb + NX;    // NX
  float* out = (float*)d_out;

  f32_to_bf16_kernel<<<(int)((NX + 255) / 256), 256, 0, stream>>>(hs, Xb, (int)NX);
  f32_to_bf16_kernel<<<(int)((NW + 255) / 256), 256, 0, stream>>>(Wq, Wqb, (int)NW);
  f32_to_bf16_kernel<<<(int)((NW + 255) / 256), 256, 0, stream>>>(Wk, Wkb, (int)NW);
  f32_to_bf16_kernel<<<(int)((NW + 255) / 256), 256, 0, stream>>>(Wv, Wvb, (int)NW);
  f32_to_bf16_kernel<<<(int)((NW + 255) / 256), 256, 0, stream>>>(Wo, Wob, (int)NW);

  dim3 gq(D_MODEL / 64, M_TOTAL / 128, 3);
  gemm_qkv_bf16<<<gq, 128, 0, stream>>>(Xb, Wqb, Wkb, Wvb, QKVb);

  dim3 ga(S_LEN / 64, NH, BATCH);
  flash_attn_bf16<<<ga, 128, 0, stream>>>(Qb, Kb, Vb, rel, Cb);

  dim3 go(D_MODEL / 64, M_TOTAL / 128);
  gemm_out_f32<<<go, 128, 0, stream>>>(Cb, Wob, out);
}
